// CTC_76922864271604
// MI455X (gfx1250) — matrix operand order repaired and checked by reading, not compiled, not run on hardware
//
#include <hip/hip_runtime.h>
#include <hip/hip_bf16.h>

// ---------------------------------------------------------------------------
// Fused CTC loss for MI455X (gfx1250, wave32, WMMA).
//   K0: W fp32 -> f16 (one-shot, makes hot-loop B-fragments 2x b128 loads)
//   K1: fused projection + streaming logsumexp + label-logit gather -> lp_ext
//       (v_wmma_f32_16x16x32_f16, fp32 accumulate; logits never hit HBM;
//        B-fragments double-buffered, A-fragments in distinct regs so all
//        six ds_load_b128 issue in one clause)
//   K2: CTC alpha forward DP (per-batch block, alpha in LDS)
//   K3: scalar reduction
// ---------------------------------------------------------------------------

typedef __attribute__((ext_vector_type(16))) _Float16 v16h;
typedef __attribute__((ext_vector_type(8)))  _Float16 v8h;
typedef __attribute__((ext_vector_type(4)))  _Float16 v4h;
typedef __attribute__((ext_vector_type(8)))  float    v8f;

namespace ctc {
constexpr int BATCH = 16;
constexpr int T     = 1000;
constexpr int D     = 512;
constexpr int V     = 5000;
constexpr int L     = 64;
constexpr int S     = 2 * L + 1;          // 129
constexpr int MROWS = 48;                 // rows (b,t) per block: 3 WMMA M-tiles
constexpr int MT    = MROWS / 16;         // 3
constexpr int LDAP  = D + 8;              // f16 pitch; 4-bank stride kills conflicts
constexpr int NT    = (V + 15) / 16;      // 313 column tiles
constexpr int LT    = (S + 15) / 16;      // 9 label column tiles
constexpr float NEGF = -1e30f;
}

__device__ __forceinline__ float lax(float x, float y) {
  float m = fmaxf(x, y);
  return m + __logf(__expf(x - m) + __expf(y - m));
}

// One K-step: 3 independent A-fragment loads (distinct regs -> one ds clause),
// then 3 WMMAs sharing the B fragment.
__device__ __forceinline__ void wmma_step3(const _Float16* __restrict__ ldsA,
                                           int rc, int half, int kb,
                                           const v16h& bf, v8f (&acc)[ctc::MT]) {
  using namespace ctc;
  const _Float16* base = ldsA + rc * LDAP + kb + half * 16;
  v16h f0, f1, f2;
#pragma unroll
  for (int i = 0; i < 8; ++i) {
    f0[i] = base[i];                 f0[8 + i] = base[8 + i];
    f1[i] = base[16 * LDAP + i];     f1[8 + i] = base[16 * LDAP + 8 + i];
    f2[i] = base[32 * LDAP + i];     f2[8 + i] = base[32 * LDAP + 8 + i];
  }
  acc[0] = __builtin_amdgcn_wmma_f32_16x16x32_f16(false, f0, false, bf, (short)0, acc[0], false, false);
  acc[1] = __builtin_amdgcn_wmma_f32_16x16x32_f16(false, f1, false, bf, (short)0, acc[1], false, false);
  acc[2] = __builtin_amdgcn_wmma_f32_16x16x32_f16(false, f2, false, bf, (short)0, acc[2], false, false);
}

__device__ __forceinline__ v16h pack16(const v8h& lo, const v8h& hi) {
  v16h r;
#pragma unroll
  for (int i = 0; i < 8; ++i) { r[i] = lo[i]; r[8 + i] = hi[i]; }
  return r;
}

// ---------------------------------------------------------------------------
// Kernel 0: W [V,D] fp32 -> f16 into workspace (one shot, bandwidth-trivial)
// ---------------------------------------------------------------------------
__global__ __launch_bounds__(256) void w_to_f16_kernel(
    const float* __restrict__ W, _Float16* __restrict__ Wh) {
  using namespace ctc;
  const size_t i = ((size_t)blockIdx.x * 256 + threadIdx.x) * 8;
  if (i < (size_t)V * D) {
    float4 a = *(const float4*)(W + i);
    float4 b = *(const float4*)(W + i + 4);
    v8h h = {(_Float16)a.x, (_Float16)a.y, (_Float16)a.z, (_Float16)a.w,
             (_Float16)b.x, (_Float16)b.y, (_Float16)b.z, (_Float16)b.w};
    *(v8h*)(Wh + i) = h;
  }
}

// ---------------------------------------------------------------------------
// Kernel 1: per (b, t-tile): flash logsumexp over V + label logits -> lp_ext
// grid = (ceil(T/48), B), block = 256 (8 waves)
// ---------------------------------------------------------------------------
__global__ __launch_bounds__(256) void ctc_logits_kernel(
    const float* __restrict__ hs,      // [B,T,D]
    const _Float16* __restrict__ Wh,   // [V,D] f16
    const float* __restrict__ bias,    // [V]
    const int*   __restrict__ ys,      // [B,L]
    float*       __restrict__ lp)      // [T,B,S] out
{
  using namespace ctc;
  __shared__ _Float16 ldsA[MROWS * LDAP];
  __shared__ float    red[8][MROWS][2];
  __shared__ float    lse[MROWS];

  const int b    = blockIdx.y;
  const int t0   = blockIdx.x * MROWS;
  const int tid  = threadIdx.x;
  const int lane = tid & 31;
  const int wave = tid >> 5;
  const int half = lane >> 4;    // which 16-lane half
  const int rc   = lane & 15;    // row/col within half

  // ---- stage hs tile (fp32 -> f16) into LDS: 48 x 512 ----
  for (int i = tid; i < MROWS * (D / 4); i += 256) {
    const int row = i >> 7;            // D/4 = 128 float4 per row
    const int c4  = (i & 127) * 4;
    const int t   = t0 + row;
    float4 v = make_float4(0.f, 0.f, 0.f, 0.f);
    if (t < T) v = *(const float4*)(hs + ((size_t)b * T + t) * D + c4);
    v4h h = { (_Float16)v.x, (_Float16)v.y, (_Float16)v.z, (_Float16)v.w };
    *(v4h*)(ldsA + row * LDAP + c4) = h;
  }
  __syncthreads();

  // ---- streaming logsumexp over all V columns (per-lane flash state) ----
  float lmax[MT][8], lsum[MT][8];
#pragma unroll
  for (int mt = 0; mt < MT; ++mt)
#pragma unroll
    for (int r = 0; r < 8; ++r) { lmax[mt][r] = NEGF; lsum[mt][r] = 0.f; }

  for (int c = wave; c < NT; c += 8) {           // uniform per wave
    const int   vcol = c * 16 + rc;
    const int   vc   = vcol < V ? vcol : V - 1;
    const float bval = vcol < V ? bias[vcol] : NEGF;
    const _Float16* wrow = Wh + (size_t)vc * D + half * 16;

    // prefetch this wave's next tile (global_prefetch_b8)
    if (vcol + 128 < V) __builtin_prefetch(wrow + (size_t)128 * D, 0, 1);

    v8f acc[MT] = {};
    // B fragment double buffer: load kb+32 while kb's WMMAs run
    v8h nb0 = *(const v8h*)(wrow);
    v8h nb1 = *(const v8h*)(wrow + 8);
#pragma unroll 4
    for (int kb = 0; kb < D - 32; kb += 32) {
      const v16h bf = pack16(nb0, nb1);
      nb0 = *(const v8h*)(wrow + kb + 32);
      nb1 = *(const v8h*)(wrow + kb + 40);
      wmma_step3(ldsA, rc, half, kb, bf, acc);
    }
    wmma_step3(ldsA, rc, half, D - 32, pack16(nb0, nb1), acc);

    // lane-local flash update (no shuffles in hot loop; exp co-executes w/ XDL)
#pragma unroll
    for (int mt = 0; mt < MT; ++mt)
#pragma unroll
      for (int r = 0; r < 8; ++r) {
        const float v  = acc[mt][r] + bval;
        const float nm = fmaxf(lmax[mt][r], v);
        lsum[mt][r] = lsum[mt][r] * __expf(lmax[mt][r] - nm) + __expf(v - nm);
        lmax[mt][r] = nm;
      }
  }

  // cross-lane merge (16 columns per half) once
#pragma unroll
  for (int mt = 0; mt < MT; ++mt)
#pragma unroll
    for (int r = 0; r < 8; ++r) {
      float m = lmax[mt][r], s = lsum[mt][r];
#pragma unroll
      for (int off = 1; off < 16; off <<= 1) {
        const float om = __shfl_xor(m, off, 32);
        const float os = __shfl_xor(s, off, 32);
        const float nm = fmaxf(m, om);
        s = s * __expf(m - nm) + os * __expf(om - nm);
        m = nm;
      }
      lmax[mt][r] = m; lsum[mt][r] = s;
    }
  if (rc == 0) {
#pragma unroll
    for (int mt = 0; mt < MT; ++mt)
#pragma unroll
      for (int r = 0; r < 8; ++r) {
        const int row = mt * 16 + half * 8 + r;
        red[wave][row][0] = lmax[mt][r];
        red[wave][row][1] = lsum[mt][r];
      }
  }
  __syncthreads();
  if (tid < MROWS) {
    float gm = NEGF;
#pragma unroll
    for (int w = 0; w < 8; ++w) gm = fmaxf(gm, red[w][tid][0]);
    float gs = 0.f;
#pragma unroll
    for (int w = 0; w < 8; ++w) gs += red[w][tid][1] * __expf(red[w][tid][0] - gm);
    lse[tid] = gm + __logf(gs);
  }
  __syncthreads();

  // ---- label pass: 9 column tiles of gathered W rows at extended labels ----
  for (int c = wave; c < LT; c += 8) {           // wave0 does c=0 and c=8
    const int s = c * 16 + rc;
    int lab = 0;
    if (s < S && (s & 1)) lab = ys[b * L + (s >> 1)];  // blank=0 on even s
    const float bval = bias[lab];
    const _Float16* wrow = Wh + (size_t)lab * D + half * 16;

    v8f acc[MT] = {};
    v8h nb0 = *(const v8h*)(wrow);
    v8h nb1 = *(const v8h*)(wrow + 8);
#pragma unroll 4
    for (int kb = 0; kb < D - 32; kb += 32) {
      const v16h bf = pack16(nb0, nb1);
      nb0 = *(const v8h*)(wrow + kb + 32);
      nb1 = *(const v8h*)(wrow + kb + 40);
      wmma_step3(ldsA, rc, half, kb, bf, acc);
    }
    wmma_step3(ldsA, rc, half, D - 32, pack16(nb0, nb1), acc);

#pragma unroll
    for (int mt = 0; mt < MT; ++mt)
#pragma unroll
      for (int r = 0; r < 8; ++r) {
        const int row = mt * 16 + half * 8 + r;
        const int t   = t0 + row;
        if (s < S && t < T) {
          lp[((size_t)t * BATCH + b) * S + s] = acc[mt][r] + bval - lse[row];
        }
      }
  }
}

// ---------------------------------------------------------------------------
// Kernel 2: CTC alpha DP. grid = B, block = 160 (s = tid), alpha in LDS.
// ---------------------------------------------------------------------------
__global__ __launch_bounds__(160) void ctc_alpha_kernel(
    const float* __restrict__ lp,     // [T,B,S]
    const int*   __restrict__ hlens,  // [B]
    const int*   __restrict__ ys,     // [B,L]
    const int*   __restrict__ ylens,  // [B]
    float*       __restrict__ loss)   // [B]
{
  using namespace ctc;
  __shared__ float alpha[S];
  const int b  = blockIdx.x;
  const int s  = threadIdx.x;
  const int hl = hlens[b];
  const int yl = ylens[b];

  bool allow = false, svalid = false;
  if (s < S) {
    const int ext   = (s & 1) ? ys[b * L + (s >> 1)] : 0;
    const int extm2 = (s >= 2 && (s & 1)) ? ys[b * L + ((s - 2) >> 1)] : 0;
    allow  = (s >= 2) && (ext != 0) && (ext != extm2);
    svalid = s < 2 * yl + 1;
    float a0 = (s <= 1) ? lp[(size_t)b * S + s] : NEGF;  // t = 0
    if (!svalid) a0 = NEGF;
    alpha[s] = a0;
  }
  __syncthreads();

  for (int t = 1; t < T; ++t) {
    const float a0 = (s < S) ? alpha[s] : NEGF;
    const float a1 = (s >= 1 && s < S) ? alpha[s - 1] : NEGF;
    const float a2 = (s >= 2 && s < S) ? alpha[s - 2] : NEGF;
    __syncthreads();
    if (s < S) {
      const float cand = lax(lax(a0, a1), allow ? a2 : NEGF);
      float nv = svalid ? cand + lp[((size_t)t * BATCH + b) * S + s] : NEGF;
      alpha[s] = (t < hl) ? nv : a0;   // freeze past sequence length
    }
    __syncthreads();
  }

  if (threadIdx.x == 0) {
    const int send = 2 * yl;
    const float a_last = alpha[send];
    const float a_prev = (yl > 0) ? alpha[send - 1] : NEGF;
    loss[b] = -lax(a_last, a_prev);
  }
}

// ---------------------------------------------------------------------------
// Kernel 3: sum(loss)/B -> scalar
// ---------------------------------------------------------------------------
__global__ void ctc_reduce_kernel(const float* __restrict__ loss, float* __restrict__ out) {
  using namespace ctc;
  if (threadIdx.x == 0) {
    float sum = 0.f;
#pragma unroll
    for (int b = 0; b < BATCH; ++b) sum += loss[b];
    out[0] = sum / (float)BATCH;
  }
}

// ---------------------------------------------------------------------------
extern "C" void kernel_launch(void* const* d_in, const int* in_sizes, int n_in,
                              void* d_out, int out_size, void* d_ws, size_t ws_size,
                              hipStream_t stream) {
  using namespace ctc;
  const float* hs    = (const float*)d_in[0];  // [B,T,D]
  const int*   hlens = (const int*)  d_in[1];  // [B]
  const int*   ys    = (const int*)  d_in[2];  // [B,L]
  const int*   ylens = (const int*)  d_in[3];  // [B]
  const float* W     = (const float*)d_in[4];  // [V,D]
  const float* bias  = (const float*)d_in[5];  // [V]
  float* out = (float*)d_out;

  // workspace layout (16B aligned sections):
  float* lp   = (float*)d_ws;                           // [T,B,S]   8,256,000 B
  float* loss = lp + (size_t)T * BATCH * S;             // [B]
  _Float16* Wh = (_Float16*)(loss + BATCH);             // [V,D] f16 5,120,000 B

  const int wblocks = (int)(((size_t)V * D / 8 + 255) / 256);
  w_to_f16_kernel<<<wblocks, 256, 0, stream>>>(W, Wh);

  dim3 gridA((T + MROWS - 1) / MROWS, BATCH);
  ctc_logits_kernel<<<gridA, 256, 0, stream>>>(hs, Wh, bias, ys, lp);
  ctc_alpha_kernel<<<dim3(BATCH), dim3(160), 0, stream>>>(lp, hlens, ys, ylens, loss);
  ctc_reduce_kernel<<<1, 32, 0, stream>>>(loss, out);
}
